// HeteroGraph_75514114998320
// MI455X (gfx1250) — matrix operand order repaired and checked by reading, not compiled, not run on hardware
//
#include <hip/hip_runtime.h>
#include <hip/hip_bf16.h>
#include <stdint.h>
#include <stddef.h>

// ---------------------------------------------------------------------------
// HGT (heterogeneous graph transformer) on MI455X / gfx1250.
// Dense 128x128 projections + per-head relation transforms run on
// v_wmma_f32_16x16x32_bf16 (bf16 in, f32 accumulate). Relation matrices
// (4 heads x 32x32) are packed block-diagonal into 128x128 so the same WMMA
// kernel serves them. Weights are staged into LDS via the Tensor Data Mover
// (tensor_load_to_lds + s_wait_tensorcnt) when the builtin is available.
// Sparse edge softmax/aggregation is atomics-based.
// ---------------------------------------------------------------------------

#define HID 128
#define NUM_GRAPHS 2048

#ifndef __has_builtin
#define __has_builtin(x) 0
#endif
#if __has_builtin(__builtin_amdgcn_tensor_load_to_lds) && \
    __has_builtin(__builtin_amdgcn_s_wait_tensorcnt)
#define HAVE_TDM 1
#else
#define HAVE_TDM 0
#endif
#if __has_include(<hip/amd_detail/amd_gfx1250_TDM.h>)
#define TDM_SIX_ARG 1   // therock-10.0 headers -> 6-arg builtin
#else
#define TDM_SIX_ARG 0   // ROCm 7.2 -> 5-arg builtin
#endif

typedef __attribute__((ext_vector_type(16))) __bf16 bf16x16;
typedef __attribute__((ext_vector_type(8)))  float  floatx8;
typedef __attribute__((ext_vector_type(4))) unsigned int uint32x4;
typedef __attribute__((ext_vector_type(8))) int int32x8;
typedef __attribute__((ext_vector_type(4))) int int32x4;

union AFrag { bf16x16 v; uint4 q[2]; };

__device__ __forceinline__ unsigned short f2bf(float f) {
  unsigned u = __float_as_uint(f);
  u += 0x7fffu + ((u >> 16) & 1u);          // round-to-nearest-even
  return (unsigned short)(u >> 16);
}
__device__ __forceinline__ float bf2f(unsigned short h) {
  return __uint_as_float(((unsigned)h) << 16);
}
// monotonic float -> uint mapping for atomicMax on floats
__device__ __forceinline__ unsigned ordOf(float f) {
  unsigned u = __float_as_uint(f);
  return (u & 0x80000000u) ? ~u : (u | 0x80000000u);
}
__device__ __forceinline__ float fromOrd(unsigned u) {
  return (u & 0x80000000u) ? __uint_as_float(u ^ 0x80000000u)
                           : __uint_as_float(~u);
}

// Stage a 128x128 bf16 weight matrix into LDS.
__device__ __forceinline__ void stage_weights(const unsigned short* __restrict__ Wt,
                                              unsigned short* __restrict__ w) {
#if HAVE_TDM
  if (threadIdx.x < 32) {                     // wave 0 issues the TDM DMA
    const unsigned long long ga = (unsigned long long)(uintptr_t)Wt;
    const unsigned ldsAddr = (unsigned)(uintptr_t)w;   // flat addr[31:0] = LDS offset
    // ---- D# group 0: count=1 | lds_addr | global_addr | type=2 ----
    uint32x4 g0;
    g0[0] = 1u;                                        // count=1, is_restore=0
    g0[1] = ldsAddr;
    g0[2] = (unsigned)(ga & 0xffffffffull);
    g0[3] = (unsigned)((ga >> 32) & 0x01ffffffull) | (2u << 30);
    // ---- D# group 1: data_size=2B, tensor 128x128, tile 128x128 ----
    int32x8 g1;
    g1[0] = (int)(1u << 16);                           // data_size = 1 (2 bytes)
    g1[1] = (int)(128u << 16);                         // tensor_dim0 lo16
    g1[2] = (int)(128u << 16);                         // tensor_dim1 lo16
    g1[3] = (int)(128u << 16);                         // tile_dim0
    g1[4] = (int)128u;                                 // tile_dim1
    g1[5] = (int)128u;                                 // tensor_dim0_stride lo32
    g1[6] = 0; g1[7] = 0;
    int32x4 gz; gz[0] = 0; gz[1] = 0; gz[2] = 0; gz[3] = 0;   // 2D: groups 2/3 unused
#if TDM_SIX_ARG
    int32x8 gz8; for (int i = 0; i < 8; ++i) gz8[i] = 0;
    __builtin_amdgcn_tensor_load_to_lds(g0, g1, gz, gz, gz8, 0);
#else
    __builtin_amdgcn_tensor_load_to_lds(g0, g1, gz, gz, 0);
#endif
    __builtin_amdgcn_s_wait_tensorcnt(0);
  }
#else
  const uint4* s = reinterpret_cast<const uint4*>(Wt);
  uint4* d = reinterpret_cast<uint4*>(w);
  for (int i = threadIdx.x; i < HID * HID / 8; i += 256) d[i] = s[i];
#endif
  __syncthreads();
}

// ---------------------------------------------------------------------------
// WMMA GEMM: D[N,128] = A_bf16[N,128] @ W + bias.  Wt is W transposed
// ([out_col][k], bf16, 128x128).  EPI=0 -> f32 out, EPI=1 -> bf16 out.
// 256 threads = 8 waves; each wave computes a 16x128 strip.
// ---------------------------------------------------------------------------
template <int EPI>
__global__ __launch_bounds__(256)
void wmma_linear(const unsigned short* __restrict__ A,
                 const unsigned short* __restrict__ Wt,
                 const float* __restrict__ bias,
                 void* __restrict__ D, int N) {
  __shared__ unsigned short w[HID * HID];          // 32 KB
  stage_weights(Wt, w);

  const int wave = threadIdx.x >> 5;
  const int lane = threadIdx.x & 31;
  const int rowBase = blockIdx.x * 128 + wave * 16;
  if (rowBase >= N) return;                        // uniform per wave
  const int laneHalf = lane >> 4;                  // 0 | 1
  const int lane15 = lane & 15;
  const int row = rowBase + lane15;
  const bool rowOK = (row < N);
  const bool fullTile = (rowBase + 16 <= N);       // uniform per wave

  // A fragments for all 4 K-steps (ISA 16-bit A 16x32 layout).
  AFrag a[4];
  const uint4* arow =
      reinterpret_cast<const uint4*>(A + (size_t)(rowOK ? row : 0) * HID);
  for (int ks = 0; ks < 4; ++ks) {
    const int e0 = ks * 4 + laneHalf;              // (ks*32 + laneHalf*8)/8
    if (rowOK) { a[ks].q[0] = arow[e0]; a[ks].q[1] = arow[e0 + 2]; }
    else { a[ks].q[0] = make_uint4(0,0,0,0); a[ks].q[1] = make_uint4(0,0,0,0); }
  }

  for (int nt = 0; nt < 8; ++nt) {
    const int n0 = nt * 16;
    floatx8 acc = {0.f,0.f,0.f,0.f,0.f,0.f,0.f,0.f};
    for (int ks = 0; ks < 4; ++ks) {
      AFrag b;                                     // B 32x16: col = lane15
      const uint4* brow = reinterpret_cast<const uint4*>(
          w + (n0 + lane15) * HID + ks * 32 + laneHalf * 8);
      b.q[0] = brow[0]; b.q[1] = brow[2];
      acc = __builtin_amdgcn_wmma_f32_16x16x32_bf16(
          false, a[ks].v, false, b.v, (short)0, acc, false, false);
    }
    const int col = n0 + lane15;
    const float bv = bias ? bias[col] : 0.f;
    const size_t base = (size_t)(rowBase + laneHalf * 8) * HID + col;
    if (fullTile) {                                // fast path: no guards
      for (int i = 0; i < 8; ++i) {
        const float val = acc[i] + bv;
        if (EPI == 0) ((float*)D)[base + (size_t)i * HID] = val;
        else ((unsigned short*)D)[base + (size_t)i * HID] = f2bf(val);
      }
    } else {
      for (int i = 0; i < 8; ++i) {
        const int r = rowBase + i + laneHalf * 8;  // C/D layout
        if (r < N) {
          const float val = acc[i] + bv;
          if (EPI == 0) ((float*)D)[(size_t)r * HID + col] = val;
          else ((unsigned short*)D)[(size_t)r * HID + col] = f2bf(val);
        }
      }
    }
  }
}

// Wt[n*128+k] = bf16(W[k*128+n])
__global__ void prep_wt(const float* __restrict__ W, unsigned short* __restrict__ Wt) {
  const int k = blockIdx.x, n = threadIdx.x;
  Wt[n * HID + k] = f2bf(W[k * HID + n]);
}
// block-diagonal relation matrix, already transposed:
// Wt[n*128+k] = rel[h][k%32][n%32] if h==n/32==k/32 else 0
__global__ void prep_bd(const float* __restrict__ rel, unsigned short* __restrict__ Wt) {
  const int k = blockIdx.x, n = threadIdx.x;
  const int hk = k >> 5, hn = n >> 5;
  const float v = (hk == hn) ? rel[(hk * 32 + (k & 31)) * 32 + (n & 31)] : 0.f;
  Wt[n * HID + k] = f2bf(v);
}

// h = x @ Win + bin  (Din <= 16), also emit bf16 copy
__global__ void in_proj(const float* __restrict__ x, const float* __restrict__ W,
                        const float* __restrict__ b, float* __restrict__ h,
                        unsigned short* __restrict__ hb, int N, int Din) {
  const int n = blockIdx.x;
  if (n >= N) return;
  const int c = threadIdx.x;
  float acc = b[c];
  for (int d = 0; d < Din; ++d) acc += x[(size_t)n * Din + d] * W[d * HID + c];
  h[(size_t)n * HID + c] = acc;
  hb[(size_t)n * HID + c] = f2bf(acc);
}

__global__ void gelu_bf16(const float* __restrict__ a, unsigned short* __restrict__ o, int n) {
  const int i = blockIdx.x * 256 + threadIdx.x;
  if (i >= n) return;
  const float x = a[i];
  const float t = 0.7978845608028654f * (x + 0.044715f * x * x * x);
  o[i] = f2bf(0.5f * x * (1.f + tanhf(t)));
}

__global__ void skip_combine(const float* __restrict__ o, float* __restrict__ h,
                             const float* __restrict__ skip, int n) {
  const int i = blockIdx.x * 256 + threadIdx.x;
  if (i >= n) return;
  const float beta = 1.f / (1.f + __expf(-skip[0]));
  h[i] = beta * o[i] + (1.f - beta) * h[i];
}

// ELU then LayerNorm over 128 channels; optionally emit bf16 copy
__global__ void elu_ln(float* __restrict__ h, unsigned short* __restrict__ hb,
                       const float* __restrict__ gamma, const float* __restrict__ beta,
                       int N, int writeBf) {
  __shared__ float red[HID];
  const int n = blockIdx.x;
  if (n >= N) return;
  const int c = threadIdx.x;
  float x = h[(size_t)n * HID + c];
  x = x > 0.f ? x : (__expf(x) - 1.f);
  red[c] = x; __syncthreads();
  for (int s = 64; s > 0; s >>= 1) { if (c < s) red[c] += red[c + s]; __syncthreads(); }
  const float mu = red[0] * (1.f / HID); __syncthreads();
  const float d = x - mu;
  red[c] = d * d; __syncthreads();
  for (int s = 64; s > 0; s >>= 1) { if (c < s) red[c] += red[c + s]; __syncthreads(); }
  const float y = d * rsqrtf(red[0] * (1.f / HID) + 1e-5f) * gamma[c] + beta[c];
  h[(size_t)n * HID + c] = y;
  if (writeBf) hb[(size_t)n * HID + c] = f2bf(y);
}

// ------------------- edge kernels (segment softmax + scatter) --------------
__global__ void edge_logits(const int* __restrict__ src, const int* __restrict__ dst,
                            const float* __restrict__ kr, const unsigned short* __restrict__ qb,
                            const float* __restrict__ p_rel, float* __restrict__ logits,
                            unsigned* __restrict__ mx, int E) {
  const int t = blockIdx.x * 256 + threadIdx.x;
  if (t >= E * 4) return;
  const int e = t >> 2, h = t & 3;
  const int s = src[e], d = dst[e];
  const float4* kp = reinterpret_cast<const float4*>(kr + (size_t)s * HID + h * 32);
  const uint4* qp = reinterpret_cast<const uint4*>(qb + (size_t)d * HID + h * 32);
  float acc = 0.f;
  for (int j = 0; j < 4; ++j) {
    const uint4 qv = qp[j];
    const float4 k0 = kp[2 * j], k1 = kp[2 * j + 1];
    acc += k0.x * bf2f((unsigned short)(qv.x)) + k0.y * bf2f((unsigned short)(qv.x >> 16));
    acc += k0.z * bf2f((unsigned short)(qv.y)) + k0.w * bf2f((unsigned short)(qv.y >> 16));
    acc += k1.x * bf2f((unsigned short)(qv.z)) + k1.y * bf2f((unsigned short)(qv.z >> 16));
    acc += k1.z * bf2f((unsigned short)(qv.w)) + k1.w * bf2f((unsigned short)(qv.w >> 16));
  }
  const float l = acc * p_rel[h] * 0.17677669529663687f;  // 1/sqrt(32)
  logits[t] = l;
  atomicMax(&mx[d * 4 + h], ordOf(l));
}

__global__ void edge_exp(const int* __restrict__ dst, float* __restrict__ logits,
                         const unsigned* __restrict__ mx, float* __restrict__ sm, int E) {
  const int t = blockIdx.x * 256 + threadIdx.x;
  if (t >= E * 4) return;
  const int e = t >> 2, h = t & 3;
  const int d = dst[e];
  float m = fromOrd(mx[d * 4 + h]);
  if (!isfinite(m)) m = 0.f;
  const float ex = __expf(logits[t] - m);
  logits[t] = ex;
  atomicAdd(&sm[d * 4 + h], ex);
}

__global__ void edge_scatter(const int* __restrict__ src, const int* __restrict__ dst,
                             const float* __restrict__ vr, const float* __restrict__ logits,
                             const float* __restrict__ sm, float* __restrict__ agg, int E) {
  const int t = blockIdx.x * 256 + threadIdx.x;
  if (t >= E * 4) return;
  const int e = t >> 2, h = t & 3;
  const int s = src[e], d = dst[e];
  const float alpha = logits[t] / (sm[d * 4 + h] + 1e-16f);
  const float4* vp = reinterpret_cast<const float4*>(vr + (size_t)s * HID + h * 32);
  float* ap = agg + (size_t)d * HID + h * 32;
  for (int j = 0; j < 8; ++j) {
    const float4 v4 = vp[j];
    atomicAdd(&ap[4 * j + 0], v4.x * alpha);
    atomicAdd(&ap[4 * j + 1], v4.y * alpha);
    atomicAdd(&ap[4 * j + 2], v4.z * alpha);
    atomicAdd(&ap[4 * j + 3], v4.w * alpha);
  }
}

// ------------------- pooling + output head --------------------------------
__global__ void pool_sum(const float* __restrict__ op, const int* __restrict__ batch,
                         float* __restrict__ sums, float* __restrict__ cnt, int N) {
  const int n = blockIdx.x;
  if (n >= N) return;
  const int c = threadIdx.x;
  const int g = batch[n];
  atomicAdd(&sums[(size_t)g * HID + c], op[(size_t)n * HID + c]);
  if (c == 0) atomicAdd(&cnt[g], 1.f);
}

__global__ void pool_out(const float* __restrict__ sums, const float* __restrict__ cnt,
                         const float* __restrict__ W, const float* __restrict__ b,
                         float* __restrict__ out) {
  __shared__ float red[HID];
  const int g = blockIdx.x, c = threadIdx.x;
  float cc = cnt[g]; cc = cc > 1.f ? cc : 1.f;
  red[c] = (sums[(size_t)g * HID + c] / cc) * W[c];
  __syncthreads();
  for (int s = 64; s > 0; s >>= 1) { if (c < s) red[c] += red[c + s]; __syncthreads(); }
  if (c == 0) out[g] = red[0] + b[0];
}

// ---------------------------------------------------------------------------
extern "C" void kernel_launch(void* const* d_in, const int* in_sizes, int n_in,
                              void* d_out, int out_size, void* d_ws, size_t ws_size,
                              hipStream_t stream) {
  (void)in_sizes; (void)n_in; (void)out_size; (void)ws_size;

  static const int NN[4]   = {100000, 30000, 150000, 80000};   // op, tab, col, pred
  static const int DIN[4]  = {4, 2, 16, 4};
  static const int ESRC[8] = {1, 3, 2, 2, 0, 1, 2, 3};
  static const int EDST[8] = {0, 0, 0, 3, 0, 1, 2, 3};
  static const int ECNT[8] = {200000, 300000, 400000, 300000, 200000, 30000, 150000, 150000};
  const int NMAX = 150000, EMAX = 400000;

  // --- deterministic bump allocator over d_ws (~0.9 GB) ---
  char* wp = (char*)d_ws;
  auto alloc = [&](size_t bytes) -> void* {
    void* r = wp; wp += (bytes + 255) & ~(size_t)255; return r;
  };
  float* h[4]; unsigned short* hb[4]; unsigned short* kb[4];
  unsigned short* vb[4]; unsigned short* qb[4]; float* agg[4];
  for (int t = 0; t < 4; ++t) h[t]   = (float*)alloc((size_t)NN[t] * HID * 4);
  for (int t = 0; t < 4; ++t) hb[t]  = (unsigned short*)alloc((size_t)NN[t] * HID * 2);
  for (int t = 0; t < 4; ++t) kb[t]  = (unsigned short*)alloc((size_t)NN[t] * HID * 2);
  for (int t = 0; t < 4; ++t) vb[t]  = (unsigned short*)alloc((size_t)NN[t] * HID * 2);
  for (int t = 0; t < 4; ++t) qb[t]  = (unsigned short*)alloc((size_t)NN[t] * HID * 2);
  for (int t = 0; t < 4; ++t) agg[t] = (float*)alloc((size_t)NN[t] * HID * 4);
  float* krf = (float*)alloc((size_t)NMAX * HID * 4);
  float* vrf = (float*)alloc((size_t)NMAX * HID * 4);
  float* logits = (float*)alloc((size_t)EMAX * 4 * 4);
  unsigned* mx = (unsigned*)alloc((size_t)NMAX * 4 * 4);
  float* sm = (float*)alloc((size_t)NMAX * 4 * 4);
  unsigned short* Wt  = (unsigned short*)alloc(HID * HID * 2);
  unsigned short* BDa = (unsigned short*)alloc(HID * HID * 2);
  unsigned short* BDm = (unsigned short*)alloc(HID * HID * 2);
  float* sums = (float*)alloc((size_t)NUM_GRAPHS * HID * 4);
  float* cnt  = (float*)alloc((size_t)NUM_GRAPHS * 4);

  auto F  = [&](int i) { return (const float*)d_in[i]; };
  auto I  = [&](int i) { return (const int*)d_in[i]; };
  auto gb = [](int n) { return (n + 127) / 128; };
  auto ge = [](int n) { return (n + 255) / 256; };

  // --- input projections: d_in[13 + 2t] = W, [14 + 2t] = b ---
  for (int t = 0; t < 4; ++t)
    in_proj<<<NN[t], 128, 0, stream>>>(F(t), F(13 + 2 * t), F(14 + 2 * t),
                                       h[t], hb[t], NN[t], DIN[t]);

  auto run_conv = [&](int base) {
    // k/q/v projections (bf16 epilogue) + zero aggregation
    for (int t = 0; t < 4; ++t) {
      prep_wt<<<128, 128, 0, stream>>>(F(base + t * 2), Wt);
      wmma_linear<1><<<gb(NN[t]), 256, 0, stream>>>(hb[t], Wt, F(base + t * 2 + 1), kb[t], NN[t]);
      prep_wt<<<128, 128, 0, stream>>>(F(base + 8 + t * 2), Wt);
      wmma_linear<1><<<gb(NN[t]), 256, 0, stream>>>(hb[t], Wt, F(base + 8 + t * 2 + 1), qb[t], NN[t]);
      prep_wt<<<128, 128, 0, stream>>>(F(base + 16 + t * 2), Wt);
      wmma_linear<1><<<gb(NN[t]), 256, 0, stream>>>(hb[t], Wt, F(base + 16 + t * 2 + 1), vb[t], NN[t]);
      hipMemsetAsync(agg[t], 0, (size_t)NN[t] * HID * 4, stream);
    }
    // per-edge-type attention
    for (int e = 0; e < 8; ++e) {
      const int s = ESRC[e], d = EDST[e], E = ECNT[e];
      const int* src = I(4 + e);
      const int* dst = src + E;
      prep_bd<<<128, 128, 0, stream>>>(F(base + 36 + e), BDa);
      prep_bd<<<128, 128, 0, stream>>>(F(base + 44 + e), BDm);
      wmma_linear<0><<<gb(NN[s]), 256, 0, stream>>>(kb[s], BDa, nullptr, krf, NN[s]);
      wmma_linear<0><<<gb(NN[s]), 256, 0, stream>>>(vb[s], BDm, nullptr, vrf, NN[s]);
      hipMemsetAsync(mx, 0, (size_t)NN[d] * 4 * 4, stream);
      hipMemsetAsync(sm, 0, (size_t)NN[d] * 4 * 4, stream);
      edge_logits<<<ge(E * 4), 256, 0, stream>>>(src, dst, krf, qb[d], F(base + 52 + e), logits, mx, E);
      edge_exp<<<ge(E * 4), 256, 0, stream>>>(dst, logits, mx, sm, E);
      edge_scatter<<<ge(E * 4), 256, 0, stream>>>(src, dst, vrf, logits, sm, agg[d], E);
    }
    // GELU -> 'a' projection -> sigmoid-skip
    for (int t = 0; t < 4; ++t) {
      gelu_bf16<<<ge(NN[t] * HID), 256, 0, stream>>>(agg[t], hb[t], NN[t] * HID);
      prep_wt<<<128, 128, 0, stream>>>(F(base + 24 + t * 2), Wt);
      wmma_linear<0><<<gb(NN[t]), 256, 0, stream>>>(hb[t], Wt, F(base + 24 + t * 2 + 1), krf, NN[t]);
      skip_combine<<<ge(NN[t] * HID), 256, 0, stream>>>(krf, h[t], F(base + 32 + t), NN[t] * HID);
    }
  };

  // conv1 (params at 21..80) then ELU+LayerNorm(norm1) for all types
  run_conv(21);
  for (int t = 0; t < 4; ++t)
    elu_ln<<<NN[t], 128, 0, stream>>>(h[t], hb[t], F(141), F(142), NN[t], 1);

  // conv2 (params at 81..140) then ELU+LayerNorm(norm2) for operator only
  run_conv(81);
  elu_ln<<<NN[0], 128, 0, stream>>>(h[0], hb[0], F(143), F(144), NN[0], 0);

  // mean-pool operator nodes per graph + final 128->1 head
  hipMemsetAsync(sums, 0, (size_t)NUM_GRAPHS * HID * 4, stream);
  hipMemsetAsync(cnt, 0, (size_t)NUM_GRAPHS * 4, stream);
  pool_sum<<<NN[0], 128, 0, stream>>>(h[0], I(12), sums, cnt, NN[0]);
  pool_out<<<NUM_GRAPHS, 128, 0, stream>>>(sums, cnt, F(145), F(146), (float*)d_out);
}
